// KANMF_35098472743378
// MI455X (gfx1250) — matrix operand order repaired and checked by reading, not compiled, not run on hardware
//
#include <hip/hip_runtime.h>
#include <hip/hip_bf16.h>
#include <cmath>

// ---------------------------------------------------------------------------
// KAN+MF fused inference for gfx1250 (MI455X).
//
// KAN layer reformulated as GEMM: A[b, i*9+g] (8 cubic B-spline basis values +
// silu(x) per input feature, fp16) times W[i*9+g, o] (coef*scale_sp with
// scale_base in the 9th slot, prepacked fp16, row-major W^T).
// GEMMs run on v_wmma_f32_16x16x32_f16 (wave32).
// ---------------------------------------------------------------------------

typedef __attribute__((ext_vector_type(16))) _Float16 v16h;
typedef __attribute__((ext_vector_type(8)))  float    v8f;
typedef __attribute__((ext_vector_type(4)))  float    f32x4;

#define ROWS 64          // batch rows per block
#define AS   648         // LDS activation row stride in f16 (648*2 B, 16B aligned, de-conflicted)
#define KP1  576         // K for 64-input layers  (64*9)
#define KS1  18          // 576/32 wmma K-steps
#define KPA  448         // K for affine layer (48*9 = 432, padded to 448)
#define KSA  14          // 448/32
#define SMEM_BYTES (ROWS*AS*2 + ROWS*64*4 + ROWS*32*4)   // sA + sY + sMF = 107520

union ABfrag { f32x4 f[2]; v16h h; };

// Uniform cubic B-spline basis (grid h=0.4 on [-1,1], k=3) + silu, 9 fp16 outputs.
__device__ __forceinline__ void write_act(_Float16* rowp, int i, float x) {
    const float u = (x + 2.2f) * 2.5f;        // (x - t0)/h, t0 = -2.2
    _Float16* p = rowp + i * 9;
#pragma unroll
    for (int m = 0; m < 8; ++m) {
        float t = u - (float)m;
        float s = fabsf(t - 2.0f);            // symmetric cardinal form
        float b = 0.0f;
        if (s < 2.0f) {
            float w = 2.0f - s;
            b = w * w * w * (1.0f / 6.0f);
            if (s < 1.0f) { float z = 1.0f - s; b -= 4.0f * z * z * z * (1.0f / 6.0f); }
        }
        p[m] = (_Float16)b;
    }
    float sig = 1.0f / (1.0f + __expf(-x));
    p[8] = (_Float16)(x * sig);               // silu slot -> scale_base column
}

// One 16x16 output tile: A from LDS (padded stride AS), B (=W^T rows) from global.
__device__ __forceinline__ void mma_tile(const _Float16* sA, int mt,
                                         const _Float16* Wt, int kp, int nt,
                                         int ksteps, v8f& acc) {
    const int lane = threadIdx.x & 31;
    const int lm = lane & 15, hh = lane >> 4;
    const _Float16* ap = sA + (mt * 16 + lm) * AS + hh * 8;    // A: row M=lm, K-chunks per layout
    const _Float16* bp = Wt + (nt * 16 + lm) * kp + hh * 8;    // B: col N=lm, K-chunks per layout
#pragma unroll
    for (int kt = 0; kt < ksteps; ++kt) {
        ABfrag a, b;
        a.f[0] = *(const f32x4*)(ap + kt * 32);        // K 0..7   (or 8..15  on hi half)
        a.f[1] = *(const f32x4*)(ap + kt * 32 + 16);   // K 16..23 (or 24..31 on hi half)
        b.f[0] = *(const f32x4*)(bp + kt * 32);
        b.f[1] = *(const f32x4*)(bp + kt * 32 + 16);
        acc = __builtin_amdgcn_wmma_f32_16x16x32_f16(false, a.h, false, b.h,
                                                     (short)0, acc, false, false);
    }
}

// Scatter a 16x16 f32 C tile to LDS per the documented C layout
// (lanes 0-15: N=lane, M=j; lanes 16-31: N=lane-16, M=j+8).
__device__ __forceinline__ void store_c(float* sY, int ldy, int mt, int nt, const v8f& acc) {
    const int lane = threadIdx.x & 31;
    const int lm = lane & 15, hh = lane >> 4;
#pragma unroll
    for (int j = 0; j < 8; ++j)
        sY[(mt * 16 + j + hh * 8) * ldy + nt * 16 + lm] = acc[j];
}

// ---------------------------------------------------------------------------
// Prep: pack weights as fp16 W^T row-major  [O][Kp]; affine padded to O=16,K=448.
// ---------------------------------------------------------------------------
__global__ void kanmf_prep(const float* __restrict__ coef1, const float* __restrict__ sb1,
                           const float* __restrict__ ss1,
                           const float* __restrict__ coef2, const float* __restrict__ sb2,
                           const float* __restrict__ ss2,
                           const float* __restrict__ coefa, const float* __restrict__ sba,
                           const float* __restrict__ ssa,
                           _Float16* __restrict__ Wt1, _Float16* __restrict__ Wt2,
                           _Float16* __restrict__ Wta) {
    const int t = blockIdx.x * blockDim.x + threadIdx.x;
    const int N1 = 64 * KP1, N2 = 16 * KP1, N3 = 16 * KPA;
    if (t < N1) {
        int o = t / KP1, k = t % KP1, i = k / 9, g = k % 9;
        float v = (g < 8) ? coef1[(i * 64 + o) * 8 + g] * ss1[i * 64 + o] : sb1[i * 64 + o];
        Wt1[o * KP1 + k] = (_Float16)v;
    } else if (t < N1 + N2) {
        int u = t - N1, o = u / KP1, k = u % KP1, i = k / 9, g = k % 9;
        float v = (g < 8) ? coef2[(i * 16 + o) * 8 + g] * ss2[i * 16 + o] : sb2[i * 16 + o];
        Wt2[o * KP1 + k] = (_Float16)v;
    } else if (t < N1 + N2 + N3) {
        int u = t - N1 - N2, o = u / KPA, k = u % KPA;
        float v = 0.0f;
        if (o == 0 && k < 432) {
            int i = k / 9, g = k % 9;
            v = (g < 8) ? coefa[i * 8 + g] * ssa[i] : sba[i];
        }
        Wta[o * KPA + k] = (_Float16)v;
    }
}

// ---------------------------------------------------------------------------
// Fused main kernel: 64 batch rows per block, 256 threads = 8 wave32.
// ---------------------------------------------------------------------------
__global__ __launch_bounds__(256) void kanmf_main(
        const int* __restrict__ uidx, const int* __restrict__ iidx,
        const float* __restrict__ uk, const float* __restrict__ ik,
        const float* __restrict__ um, const float* __restrict__ im,
        const _Float16* __restrict__ Wt1, const _Float16* __restrict__ Wt2,
        const _Float16* __restrict__ Wta,
        float* __restrict__ out, int Btot) {
    extern __shared__ char smem[];
    _Float16* sA  = (_Float16*)smem;                               // [64][AS] fp16 activations
    float*    sY  = (float*)(smem + (size_t)ROWS * AS * 2);        // [64][64] f32 layer outputs
    float*    sMF = sY + ROWS * 64;                                // [64][32] f32 mf term

    const int tid  = threadIdx.x;
    const int wave = tid >> 5;
    const int lane = tid & 31;
    const int lm = lane & 15, hh = lane >> 4;
    const int r0 = blockIdx.x * ROWS;

    // ---- stage 1: gather embeddings, spline-activate x=[uk,ik], compute mf ----
#pragma unroll
    for (int it = 0; it < ROWS * 64 / 256; ++it) {
        int idx = it * 256 + tid;
        int r = idx >> 6, i = idx & 63;
        int g = r0 + r; g = (g < Btot) ? g : (Btot - 1);
        float x = (i < 32) ? uk[(size_t)uidx[g] * 32 + i]
                           : ik[(size_t)iidx[g] * 32 + (i - 32)];
        write_act(sA + r * AS, i, x);
    }
#pragma unroll
    for (int it = 0; it < ROWS * 32 / 256; ++it) {
        int idx = it * 256 + tid;
        int r = idx >> 5, j = idx & 31;
        int g = r0 + r; g = (g < Btot) ? g : (Btot - 1);
        sMF[r * 32 + j] = um[(size_t)uidx[g] * 32 + j] * im[(size_t)iidx[g] * 32 + j];
    }
    __syncthreads();

    // ---- layer 1 GEMM: [64 x 576] x [576 x 64] -> sY ----
    for (int job = wave; job < 16; job += 8) {
        int mt = job >> 2, nt = job & 3;
        v8f acc = {};
        mma_tile(sA, mt, Wt1, KP1, nt, KS1, acc);
        store_c(sY, 64, mt, nt, acc);
    }
    __syncthreads();

    // ---- stage 2: spline-activate layer1 output (64 features) ----
#pragma unroll
    for (int it = 0; it < ROWS * 64 / 256; ++it) {
        int idx = it * 256 + tid;
        int r = idx >> 6, i = idx & 63;
        write_act(sA + r * AS, i, sY[r * 64 + i]);
    }
    __syncthreads();

    // ---- layer 2 GEMM: [64 x 576] x [576 x 16] -> sY cols 0..15 ----
    if (wave < 4) {
        v8f acc = {};
        mma_tile(sA, wave, Wt2, KP1, 0, KS1, acc);
        store_c(sY, 64, wave, 0, acc);
    }
    __syncthreads();

    // ---- stage 3: activate v=[y2, mf] (48 features), zero-pad K to 448 ----
    for (int it = 0; it < ROWS * 48 / 256; ++it) {
        int idx = it * 256 + tid;
        int r = idx / 48, i = idx % 48;
        float x = (i < 16) ? sY[r * 64 + i] : sMF[r * 32 + (i - 16)];
        write_act(sA + r * AS, i, x);
    }
#pragma unroll
    for (int it = 0; it < ROWS * 16 / 256; ++it) {
        int idx = it * 256 + tid;
        int r = idx >> 4, j = idx & 15;
        sA[r * AS + 432 + j] = (_Float16)0.0f;
    }
    __syncthreads();

    // ---- affine GEMM [64 x 448] x [448 x 16] (col 0 live) -> sigmoid -> out ----
    if (wave < 4) {
        v8f acc = {};
        mma_tile(sA, wave, Wta, KPA, 0, KSA, acc);
        if (lm == 0) {   // lanes holding N==0
#pragma unroll
            for (int j = 0; j < 8; ++j) {
                int row = wave * 16 + j + hh * 8;
                int g = r0 + row;
                if (g < Btot) out[g] = 1.0f / (1.0f + __expf(-acc[j]));
            }
        }
    }
}

// ---------------------------------------------------------------------------
extern "C" void kernel_launch(void* const* d_in, const int* in_sizes, int n_in,
                              void* d_out, int out_size, void* d_ws, size_t ws_size,
                              hipStream_t stream) {
    const int*   uidx  = (const int*)  d_in[0];
    const int*   iidx  = (const int*)  d_in[1];
    const float* uk    = (const float*)d_in[2];
    const float* ik    = (const float*)d_in[3];
    const float* um    = (const float*)d_in[4];
    const float* im    = (const float*)d_in[5];
    // d_in[6] = grid1 (uniform, recomputed analytically)
    const float* coef1 = (const float*)d_in[7];
    const float* sb1   = (const float*)d_in[8];
    const float* ss1   = (const float*)d_in[9];
    // d_in[10] = grid2
    const float* coef2 = (const float*)d_in[11];
    const float* sb2   = (const float*)d_in[12];
    const float* ss2   = (const float*)d_in[13];
    // d_in[14] = grid_a
    const float* coefa = (const float*)d_in[15];
    const float* sba   = (const float*)d_in[16];
    const float* ssa   = (const float*)d_in[17];

    const int B = in_sizes[0];
    float* out = (float*)d_out;

    _Float16* Wt1 = (_Float16*)d_ws;           // 64*576 fp16
    _Float16* Wt2 = Wt1 + 64 * KP1;            // 16*576 fp16
    _Float16* Wta = Wt2 + 16 * KP1;            // 16*448 fp16

    const int nprep = 64 * KP1 + 16 * KP1 + 16 * KPA;
    kanmf_prep<<<(nprep + 255) / 256, 256, 0, stream>>>(
        coef1, sb1, ss1, coef2, sb2, ss2, coefa, sba, ssa, Wt1, Wt2, Wta);

    (void)hipFuncSetAttribute(reinterpret_cast<const void*>(kanmf_main),
                              hipFuncAttributeMaxDynamicSharedMemorySize, SMEM_BYTES);

    const int nb = (B + ROWS - 1) / ROWS;
    kanmf_main<<<nb, 256, SMEM_BYTES, stream>>>(
        uidx, iidx, uk, ik, um, im, Wt1, Wt2, Wta, out, B);
}
